// GatedDeltaProductEncoder_11751030521974
// MI455X (gfx1250) — matrix operand order repaired and checked by reading, not compiled, not run on hardware
//
#include <hip/hip_runtime.h>

// ---------------------------------------------------------------------------
// GatedDeltaProduct encoder for MI455X (gfx1250, wave32, WMMA + TDM).
//   1. rmsnorm(x)                                   -> xn
//   2. xn @ Wq/Wk/Wv  (WMMA f16-in/f32-acc, TDM-staged B tiles)
//   3. beta = 2*sigmoid(xn@Wb), g = -exp(A)*softplus(xn@Wa+dt)
//   4. causal depthwise conv(K=4) + SiLU            -> q,k,v
//   5. l2norm(q)*scale, l2norm(k)
//   6. sequential delta-rule scan, S[128x128] per (b,h) in VGPRs
//   7. rmsnorm(o) @ Wo + x                          -> h   (fused residual)
//   8. rmsnorm(h) @ Wg, SwiGLU, @ Wd + h            -> out (fused residual)
// ---------------------------------------------------------------------------

#define BB   2
#define TT   1024
#define DD   512
#define HH   4
#define DKK  128
#define DVV  128
#define NHH  2
#define IMLP 256
#define MROWS (BB*TT)          // 2048
#define QSCALE 0.08838834764831845f   // 128^-0.5

typedef __attribute__((ext_vector_type(16))) _Float16     v16h;
typedef __attribute__((ext_vector_type(8)))  float        v8f;
typedef __attribute__((ext_vector_type(4)))  float        v4f;
typedef __attribute__((ext_vector_type(4)))  unsigned int u32x4;
typedef __attribute__((ext_vector_type(8)))  int          i32x8;
typedef __attribute__((ext_vector_type(4)))  int          i32x4;

#ifndef __has_builtin
#define __has_builtin(x) 0
#endif
#if __has_builtin(__builtin_amdgcn_tensor_load_to_lds)
#define HAVE_TDM 1
#else
#define HAVE_TDM 0
#endif

#define DEV static __device__ __forceinline__

DEV float sigmoidf_(float x) { return 1.0f / (1.0f + __expf(-x)); }
DEV float siluf_(float x)    { return x * sigmoidf_(x); }
DEV float softplusf_(float x){ return (x > 20.0f) ? x : log1pf(__expf(x)); }

DEV float wave_reduce_sum(float v) {
  #pragma unroll
  for (int m = 16; m >= 1; m >>= 1) v += __shfl_xor(v, m, 32);
  return v;
}

template <int NW>
DEV void wait_tensorcnt_imm() {
#if __has_builtin(__builtin_amdgcn_s_wait_tensorcnt)
  __builtin_amdgcn_s_wait_tensorcnt((unsigned short)NW);
#else
  asm volatile("s_wait_tensorcnt %0" :: "i"(NW) : "memory");
#endif
  asm volatile("" ::: "memory");   // keep LDS reads below the wait
}

#if HAVE_TDM
// Issue a TDM 2D-tile load: 32 rows x 64 cols of fp32 from W[K,N] row-major
// (tile start = gsrc, row stride = N elements) into LDS at lds_byte_off,
// packed contiguously (64 floats per row).
DEV void tdm_load_b_tile(const float* gsrc, unsigned lds_byte_off, int N_elems,
                         int K_elems) {
  const unsigned long long ga = (unsigned long long)(size_t)gsrc;
  u32x4 g0;
  g0.x = 1u;                                            // count=1, user desc
  g0.y = lds_byte_off;                                  // LDS dest (bytes)
  g0.z = (unsigned)(ga & 0xFFFFFFFFu);                  // global_addr[31:0]
  g0.w = (unsigned)((ga >> 32) & 0x01FFFFFFu)           // global_addr[56:32]
       | 0x80000000u;                                   // type = 2 ("image")
  i32x8 g1;
  g1[0] = (int)(2u << 16);                              // data_size = 4 bytes
  g1[1] = (int)((unsigned)(N_elems & 0xFFFF) << 16);    // tensor_dim0[15:0]
  g1[2] = (int)(((unsigned)N_elems >> 16) |             // tensor_dim0[31:16]
                ((unsigned)(K_elems & 0xFFFF) << 16));  // tensor_dim1[15:0]
  g1[3] = (int)(((unsigned)K_elems >> 16) |             // tensor_dim1[31:16]
                (64u << 16));                           // tile_dim0 = 64
  g1[4] = 32;                                           // tile_dim1=32, dim2=0
  g1[5] = N_elems;                                      // dim0_stride[31:0]
  g1[6] = 0;                                            // dim0_stride[47:32]
  g1[7] = 0;
  const i32x4 z4 = {0, 0, 0, 0};
#if __clang_major__ >= 23
  const i32x8 z8 = {0, 0, 0, 0, 0, 0, 0, 0};
  __builtin_amdgcn_tensor_load_to_lds(g0, g1, z4, z4, z8, 0);
#else
  __builtin_amdgcn_tensor_load_to_lds(g0, g1, z4, z4, 0);
#endif
}
#endif  // HAVE_TDM

// ---------------- RMSNorm over rows of 512 (one wave per row) --------------
__global__ void rmsnorm512_kernel(const float* __restrict__ x,
                                  const float* __restrict__ w,
                                  float* __restrict__ y) {
  const int row  = blockIdx.x;
  const int lane = threadIdx.x;
  const float* xr = x + (size_t)row * DD;
  float vals[16];
  float ss = 0.0f;
  #pragma unroll
  for (int i = 0; i < 16; ++i) {
    float t = xr[lane + i * 32];
    vals[i] = t;
    ss += t * t;
  }
  ss = wave_reduce_sum(ss);
  const float rs = rsqrtf(ss * (1.0f / (float)DD) + 1e-6f);
  #pragma unroll
  for (int i = 0; i < 16; ++i) {
    const int c = lane + i * 32;
    y[(size_t)row * DD + c] = vals[i] * rs * w[c];
  }
}

// ------------- WMMA GEMM: C[M,N] = A[M,K] @ W[K,N] (+ residual) ------------
// One wave per 16x64 output tile: 4 accumulators share one A fragment.
// B tiles (32x64) staged through LDS by the Tensor Data Mover, double-buffered
// and synchronized with TENSORcnt. K is compile-time -> fully unrolled.
template <int K, bool RES>
__global__ void __launch_bounds__(32)
wmma_gemm_kernel(const float* __restrict__ A, const float* __restrict__ W,
                 const float* __restrict__ res, float* __restrict__ C, int N) {
  const int n0   = blockIdx.x * 64;
  const int tm   = blockIdx.y;
  const int lane = threadIdx.x;         // 0..31 (wave32)
  const int half = lane >> 4;           // 0 | 1
  const int l15  = lane & 15;

  const float* Arow = A + (size_t)(tm * 16 + l15) * K;

#if HAVE_TDM
  // Only __shared__ object in this kernel: allocated at LDS offset 0.
  __shared__ float btile[2][32 * 64];
  tdm_load_b_tile(W + (size_t)0 * N + n0, 0u, N, K);
#endif

  v8f acc[4] = {};
  constexpr int NT = K / 32;
  #pragma unroll
  for (int kt = 0; kt < NT; ++kt) {
    const int k0 = kt * 32;
#if HAVE_TDM
    if (kt + 1 < NT) {
      tdm_load_b_tile(W + (size_t)(k0 + 32) * N + n0,
                      ((unsigned)(kt + 1) & 1u) * (32u * 64u * 4u), N, K);
      wait_tensorcnt_imm<1>();
    } else {
      wait_tensorcnt_imm<0>();
    }
#endif
    // A fragment (16x32 f16): lanes 0-15 K-halves {0..7,16..23},
    // lanes 16-31 K-halves {8..15,24..31}; two b128 loads per half.
    const int klo = half * 8;
    const v4f a0 = *(const v4f*)(Arow + k0 + klo);
    const v4f a1 = *(const v4f*)(Arow + k0 + klo + 4);
    const v4f a2 = *(const v4f*)(Arow + k0 + klo + 16);
    const v4f a3 = *(const v4f*)(Arow + k0 + klo + 20);
    v16h a;
    #pragma unroll
    for (int i = 0; i < 4; ++i) {
      a[i]      = (_Float16)a0[i];
      a[i + 4]  = (_Float16)a1[i];
      a[i + 8]  = (_Float16)a2[i];
      a[i + 12] = (_Float16)a3[i];
    }
    #pragma unroll
    for (int j = 0; j < 4; ++j) {
      // B fragment (32x16 f16): lane = column, lanes 0-15 K={0..7,16..23}, 16-31 K={8..15,24..31}.
      v16h b;
      #pragma unroll
      for (int i = 0; i < 16; ++i) {
#if HAVE_TDM
        b[i] = (_Float16)btile[kt & 1][(half * 8 + (i & 7) + ((i >> 3) << 4)) * 64 + j * 16 + l15];
#else
        b[i] = (_Float16)W[(size_t)(k0 + half * 8 + (i & 7) + ((i >> 3) << 4)) * N + n0 + j * 16 + l15];
#endif
      }
      acc[j] = __builtin_amdgcn_wmma_f32_16x16x32_f16(false, a, false, b,
                                                      (short)0, acc[j], false,
                                                      false);
    }
  }

  // C/D layout: VGPR r -> row r + 8*half, col = lane&15.
  const int crow0 = tm * 16 + half * 8;
  #pragma unroll
  for (int j = 0; j < 4; ++j) {
    const int col = n0 + j * 16 + l15;
    #pragma unroll
    for (int r = 0; r < 8; ++r) {
      const size_t idx = (size_t)(crow0 + r) * N + col;
      if (RES) C[idx] = acc[j][r] + res[idx];
      else     C[idx] = acc[j][r];
    }
  }
}

// ----------- beta = 2*sigmoid(xn @ Wb), column-per-thread (N=8) ------------
__global__ void proj_beta_kernel(const float* __restrict__ xn,
                                 const float* __restrict__ Wb,
                                 float* __restrict__ beta) {
  const int idx = blockIdx.x * blockDim.x + threadIdx.x;
  if (idx >= MROWS * NHH * HH) return;
  const int n   = idx & 7;
  const int row = idx >> 3;
  const float* xr = xn + (size_t)row * DD;
  float acc = 0.0f;
  for (int k = 0; k < DD; ++k) acc += xr[k] * Wb[(size_t)k * (NHH * HH) + n];
  beta[idx] = 2.0f * sigmoidf_(acc);
}

// ------- g = -exp(A_log) * softplus(xn @ Wa + dt_bias), N = H = 4 ----------
__global__ void proj_g_kernel(const float* __restrict__ xn,
                              const float* __restrict__ Wa,
                              const float* __restrict__ A_log,
                              const float* __restrict__ dt_bias,
                              float* __restrict__ g) {
  const int idx = blockIdx.x * blockDim.x + threadIdx.x;
  if (idx >= MROWS * HH) return;
  const int h   = idx & 3;
  const int row = idx >> 2;
  const float* xr = xn + (size_t)row * DD;
  float acc = 0.0f;
  for (int k = 0; k < DD; ++k) acc += xr[k] * Wa[(size_t)k * HH + h];
  g[idx] = -__expf(A_log[h]) * softplusf_(acc + dt_bias[h]);
}

// --------- causal depthwise conv1d (K=4, zero pad left) + SiLU -------------
__global__ void conv_silu_kernel(const float* __restrict__ xin,
                                 const float* __restrict__ w,
                                 float* __restrict__ y,
                                 int C, int total) {
  const int idx = blockIdx.x * blockDim.x + threadIdx.x;
  if (idx >= total) return;
  const int c  = idx % C;
  const int bt = idx / C;
  const int t  = bt % TT;
  float acc = 0.0f;
  #pragma unroll
  for (int j = 0; j < 4; ++j) {
    const int ts = t + j - 3;
    const float xv = (ts >= 0) ? xin[idx + (size_t)(j - 3) * C] : 0.0f;
    acc += xv * w[c * 4 + j];
  }
  y[idx] = siluf_(acc);
}

// --------------- in-place L2-norm over groups of 128 (*scale) --------------
__global__ void l2norm128_kernel(float* __restrict__ q, float scale) {
  const int gidx = blockIdx.x;
  const int lane = threadIdx.x;
  float* gp = q + (size_t)gidx * 128;
  float v[4];
  float ss = 0.0f;
  #pragma unroll
  for (int i = 0; i < 4; ++i) { v[i] = gp[lane + i * 32]; ss += v[i] * v[i]; }
  ss = wave_reduce_sum(ss);
  const float rs = rsqrtf(ss + 1e-6f) * scale;
  #pragma unroll
  for (int i = 0; i < 4; ++i) gp[lane + i * 32] = v[i] * rs;
}

// ------------- RMSNorm over groups of 128 with weight (o_norm) -------------
__global__ void rmsnorm128w_kernel(const float* __restrict__ xin,
                                   const float* __restrict__ w,
                                   float* __restrict__ y) {
  const int gidx = blockIdx.x;
  const int lane = threadIdx.x;
  const float* gp = xin + (size_t)gidx * 128;
  float v[4];
  float ss = 0.0f;
  #pragma unroll
  for (int i = 0; i < 4; ++i) { v[i] = gp[lane + i * 32]; ss += v[i] * v[i]; }
  ss = wave_reduce_sum(ss);
  const float rs = rsqrtf(ss * (1.0f / 128.0f) + 1e-6f);
  #pragma unroll
  for (int i = 0; i < 4; ++i) {
    const int c = lane + i * 32;
    y[(size_t)gidx * 128 + c] = v[i] * rs * w[c];
  }
}

// --- sequential gated delta-rule scan: one block per (b,h), S in VGPRs -----
// 128 threads; thread dv owns column S[:,dv] (128 fp32 registers).
__global__ void __launch_bounds__(128)
scan_kernel(const float* __restrict__ q, const float* __restrict__ k,
            const float* __restrict__ v, const float* __restrict__ beta,
            const float* __restrict__ g, const float* __restrict__ S0,
            float* __restrict__ o, float* __restrict__ S_last) {
  const int b  = blockIdx.x >> 2;       // H = 4
  const int h  = blockIdx.x & 3;
  const int dv = threadIdx.x;           // 0..127

  __shared__ float sq[128], sk[2][128], sv[2][128], sb[2], sg;

  float S[128];
  const float* S0p = S0 + ((size_t)(b * HH + h) * DKK) * DVV;
  #pragma unroll
  for (int dk = 0; dk < 128; ++dk) S[dk] = S0p[(size_t)dk * DVV + dv];

  for (int t = 0; t < TT; ++t) {
    const size_t bt = (size_t)b * TT + t;
    sq[dv]    = q[(bt * HH + h) * DKK + dv];
    sk[0][dv] = k[((bt * NHH + 0) * HH + h) * DKK + dv];
    sk[1][dv] = k[((bt * NHH + 1) * HH + h) * DKK + dv];
    sv[0][dv] = v[((bt * NHH + 0) * HH + h) * DVV + dv];
    sv[1][dv] = v[((bt * NHH + 1) * HH + h) * DVV + dv];
    if (dv == 0) sg = g[bt * HH + h];
    if (dv < 2)  sb[dv] = beta[(bt * NHH + dv) * HH + h];
    if (t + 1 < TT) {  // prefetch next token's operands into GL2
      __builtin_prefetch(&q[((bt + 1) * HH + h) * DKK + dv], 0, 1);
      __builtin_prefetch(&k[(((bt + 1) * NHH) * HH + h) * DKK + dv], 0, 1);
      __builtin_prefetch(&v[(((bt + 1) * NHH) * HH + h) * DVV + dv], 0, 1);
    }
    __syncthreads();

    const float eg = __expf(sg);
    #pragma unroll
    for (int dk = 0; dk < 128; ++dk) S[dk] *= eg;

    #pragma unroll
    for (int i = 0; i < 2; ++i) {           // Householder sub-steps
      float dot = 0.0f;
      #pragma unroll
      for (int dk = 0; dk < 128; ++dk) dot += sk[i][dk] * S[dk];
      const float coef = sb[i] * (sv[i][dv] - dot);
      #pragma unroll
      for (int dk = 0; dk < 128; ++dk) S[dk] += sk[i][dk] * coef;
    }

    float ov = 0.0f;
    #pragma unroll
    for (int dk = 0; dk < 128; ++dk) ov += sq[dk] * S[dk];
    o[(bt * HH + h) * DVV + dv] = ov;
    __syncthreads();
  }

  float* Sl = S_last + ((size_t)(b * HH + h) * DKK) * DVV;
  #pragma unroll
  for (int dk = 0; dk < 128; ++dk) Sl[(size_t)dk * DVV + dv] = S[dk];
}

// ---------------------- SwiGLU: u = silu(gate) * y -------------------------
__global__ void swiglu_kernel(const float* __restrict__ gbuf,
                              float* __restrict__ u) {
  const int idx = blockIdx.x * blockDim.x + threadIdx.x;
  if (idx >= MROWS * IMLP) return;
  const int i   = idx % IMLP;
  const int row = idx / IMLP;
  const float gate = gbuf[(size_t)row * (2 * IMLP) + i];
  const float yv   = gbuf[(size_t)row * (2 * IMLP) + IMLP + i];
  u[idx] = siluf_(gate) * yv;
}

// ---------------------------------------------------------------------------
extern "C" void kernel_launch(void* const* d_in, const int* in_sizes, int n_in,
                              void* d_out, int out_size, void* d_ws, size_t ws_size,
                              hipStream_t stream) {
  (void)in_sizes; (void)n_in; (void)out_size; (void)ws_size;
  const float* x       = (const float*)d_in[0];
  const float* S0      = (const float*)d_in[1];
  const float* Wq      = (const float*)d_in[2];
  const float* Wk      = (const float*)d_in[3];
  const float* Wv      = (const float*)d_in[4];
  const float* Wb      = (const float*)d_in[5];
  const float* Wa      = (const float*)d_in[6];
  const float* A_log   = (const float*)d_in[7];
  const float* dt_bias = (const float*)d_in[8];
  const float* conv_q  = (const float*)d_in[9];
  const float* conv_k  = (const float*)d_in[10];
  const float* conv_v  = (const float*)d_in[11];
  const float* o_norm  = (const float*)d_in[12];
  const float* Wo      = (const float*)d_in[13];
  const float* an_w    = (const float*)d_in[14];
  const float* mn_w    = (const float*)d_in[15];
  const float* Wg      = (const float*)d_in[16];
  const float* Wd      = (const float*)d_in[17];

  float* out    = (float*)d_out;                    // [2048, 512]
  float* S_last = out + (size_t)MROWS * DD;         // [2, 4, 128, 128]

  // workspace carve-up (~70 MB of fp32)
  float* ws = (float*)d_ws;
  size_t off = 0;
  auto alloc = [&](size_t n) { float* p = ws + off; off += n; return p; };
  float* xn    = alloc((size_t)MROWS * DD);
  float* qraw  = alloc((size_t)MROWS * (HH * DKK));
  float* kraw  = alloc((size_t)MROWS * (NHH * HH * DKK));
  float* vraw  = alloc((size_t)MROWS * (NHH * HH * DVV));
  float* qb    = alloc((size_t)MROWS * (HH * DKK));
  float* kb    = alloc((size_t)MROWS * (NHH * HH * DKK));
  float* vb    = alloc((size_t)MROWS * (NHH * HH * DVV));
  float* beta  = alloc((size_t)MROWS * (NHH * HH));
  float* gdec  = alloc((size_t)MROWS * HH);
  float* obuf  = alloc((size_t)MROWS * DD);
  float* onbuf = alloc((size_t)MROWS * DD);
  float* hbuf  = alloc((size_t)MROWS * DD);
  float* hnbuf = alloc((size_t)MROWS * DD);
  float* gbuf  = alloc((size_t)MROWS * (2 * IMLP));
  float* ubuf  = alloc((size_t)MROWS * IMLP);

  // 1. attention-branch RMSNorm
  rmsnorm512_kernel<<<MROWS, 32, 0, stream>>>(x, an_w, xn);

  // 2. q/k/v projections (WMMA, 16x64 tiles per wave)
  wmma_gemm_kernel<DD, false><<<dim3(512 / 64, MROWS / 16), 32, 0, stream>>>(
      xn, Wq, nullptr, qraw, 512);
  wmma_gemm_kernel<DD, false><<<dim3(1024 / 64, MROWS / 16), 32, 0, stream>>>(
      xn, Wk, nullptr, kraw, 1024);
  wmma_gemm_kernel<DD, false><<<dim3(1024 / 64, MROWS / 16), 32, 0, stream>>>(
      xn, Wv, nullptr, vraw, 1024);

  // 3. beta / gate scalars
  proj_beta_kernel<<<(MROWS * NHH * HH + 255) / 256, 256, 0, stream>>>(xn, Wb, beta);
  proj_g_kernel<<<(MROWS * HH + 255) / 256, 256, 0, stream>>>(xn, Wa, A_log, dt_bias, gdec);

  // 4. short conv + SiLU
  conv_silu_kernel<<<(MROWS * 512 + 255) / 256, 256, 0, stream>>>(qraw, conv_q, qb, 512, MROWS * 512);
  conv_silu_kernel<<<(MROWS * 1024 + 255) / 256, 256, 0, stream>>>(kraw, conv_k, kb, 1024, MROWS * 1024);
  conv_silu_kernel<<<(MROWS * 1024 + 255) / 256, 256, 0, stream>>>(vraw, conv_v, vb, 1024, MROWS * 1024);

  // 5. L2 norms
  l2norm128_kernel<<<MROWS * HH, 32, 0, stream>>>(qb, QSCALE);
  l2norm128_kernel<<<MROWS * NHH * HH, 32, 0, stream>>>(kb, 1.0f);

  // 6. sequential delta-rule scan (writes o and final state)
  scan_kernel<<<BB * HH, 128, 0, stream>>>(qb, kb, vb, beta, gdec, S0, obuf, S_last);

  // 7. output norm + Wo projection with residual x  ->  h
  rmsnorm128w_kernel<<<MROWS * HH, 32, 0, stream>>>(obuf, o_norm, onbuf);
  wmma_gemm_kernel<DD, true><<<dim3(512 / 64, MROWS / 16), 32, 0, stream>>>(
      onbuf, Wo, x, hbuf, 512);

  // 8. MLP: rmsnorm -> Wg -> SwiGLU -> Wd + h
  rmsnorm512_kernel<<<MROWS, 32, 0, stream>>>(hbuf, mn_w, hnbuf);
  wmma_gemm_kernel<DD, false><<<dim3((2 * IMLP) / 64, MROWS / 16), 32, 0, stream>>>(
      hnbuf, Wg, nullptr, gbuf, 2 * IMLP);
  swiglu_kernel<<<(MROWS * IMLP + 255) / 256, 256, 0, stream>>>(gbuf, ubuf);
  wmma_gemm_kernel<IMLP, true><<<dim3(512 / 64, MROWS / 16), 32, 0, stream>>>(
      ubuf, Wd, hbuf, out, 512);
}
